// qfc_12446815224546
// MI455X (gfx1250) — matrix operand order repaired and checked by reading, not compiled, not run on hardware
//
#include <hip/hip_runtime.h>

typedef __attribute__((ext_vector_type(16))) _Float16 v16h;
typedef __attribute__((ext_vector_type(8)))  _Float16 v8h;
typedef __attribute__((ext_vector_type(4)))  _Float16 v4h;
typedef __attribute__((ext_vector_type(8)))  float    v8f;

#define BM 128      // C-tile rows per workgroup
#define BN 128      // C-tile cols per workgroup
#define BK 64       // K depth per LDS stage (2 WMMA k-steps of 32)
#define NTHREADS 512

// WAGE Quantize.W, BITS_W=2: clip to [-0.5,0.5], round to {-0.5,0,0.5}.
// rintf = round-half-even, matching jnp.round.
__device__ __forceinline__ float quantize_w(float w) {
    float xc = fminf(fmaxf(w, -0.5f), 0.5f);
    return rintf(xc + xc) * 0.5f;
}

__global__ __launch_bounds__(NTHREADS)
void wage_qgemm_f16_wmma(const float* __restrict__ A,   // [M,K] fp32 row-major
                         const float* __restrict__ W,   // [K,N] fp32 row-major
                         float* __restrict__ C,         // [M,N] fp32 row-major
                         int M, int N, int K)
{
    // Fragment-ready LDS layout: [k-group of 8][row-or-col][8 halves]
    __shared__ _Float16 sA[2][BK / 8][BM][8];   // 32 KB
    __shared__ _Float16 sB[2][BK / 8][BN][8];   // 32 KB

    const int tid  = threadIdx.x;
    const int lane = tid & 31;          // wave32
    const int wave = tid >> 5;          // 0..15
    const int lh   = lane >> 4;         // lane half (0/1)
    const int lr   = lane & 15;

    const int blockN = blockIdx.x;
    const int blockM = blockIdx.y;

    // 16 waves -> 4 (M) x 4 (N); each wave computes 32x32 of the block tile
    const int m_base = (wave >> 2) * 32;
    const int n_base = (wave & 3) * 32;

    // ---- per-thread loader geometry, rebased once ----
    // A: 16 float4 spans along k (64), 32 rows per pass, 4 passes
    const int kqA = (tid & 15) * 4;
    const int mrA = tid >> 4;                               // 0..31
    const float* Abase = A + (size_t)(blockM * BM + mrA) * K + kqA;
    // B: one column, 16 consecutive k (lane-contiguous along n -> 128B/instr)
    const int nB  = tid & 127;
    const int kgB = tid >> 7;                               // 0..3
    const float* Wbase = W + (size_t)(kgB * 16) * N + (size_t)(blockN * BN + nB);

    float4 ra[4];
    float  rb[16];

    // Phase 1: issue all global loads for a stage (no waits interleaved)
    auto load_regs = [&](int k0) {
        #pragma unroll
        for (int i = 0; i < 4; ++i)
            ra[i] = *(const float4*)(Abase + (size_t)i * 32 * K + k0);
        const float* wp = Wbase + (size_t)k0 * N;
        #pragma unroll
        for (int kk = 0; kk < 16; ++kk)
            rb[kk] = wp[(size_t)kk * N];
    };

    // Phase 2: convert/quantize and store to LDS (runs after compute overlap)
    auto store_lds = [&](int s) {
        #pragma unroll
        for (int i = 0; i < 4; ++i) {
            v4h h = { (_Float16)ra[i].x, (_Float16)ra[i].y,
                      (_Float16)ra[i].z, (_Float16)ra[i].w };
            *(v4h*)&sA[s][kqA >> 3][mrA + i * 32][kqA & 7] = h;  // ds_store_b64
        }
        v8h lo, hi;
        #pragma unroll
        for (int i = 0; i < 8; ++i) {
            lo[i] = (_Float16)quantize_w(rb[i]);
            hi[i] = (_Float16)quantize_w(rb[i + 8]);
        }
        *(v8h*)&sB[s][kgB * 2    ][nB][0] = lo;                  // ds_store_b128
        *(v8h*)&sB[s][kgB * 2 + 1][nB][0] = hi;
    };

    v8f acc[2][2] = {};   // fp32 accumulators, 2x2 WMMA 16x16 tiles

    auto compute_stage = [&](int s) {
        #pragma unroll
        for (int ks = 0; ks < 2; ++ks) {
            v16h af[2], bf[2];
            // A frag (16x32 f16): lane row, k = {lh*8+0..7, 16+lh*8+0..7}
            {
                const int g0 = ks * 4 + lh;
                const int g1 = g0 + 2;
                #pragma unroll
                for (int t = 0; t < 2; ++t) {
                    const int row = m_base + t * 16 + lr;
                    v8h lo = *(const v8h*)&sA[s][g0][row][0];    // ds_load_b128
                    v8h hi = *(const v8h*)&sA[s][g1][row][0];
                    af[t] = __builtin_shufflevector(lo, hi,
                        0,1,2,3,4,5,6,7,8,9,10,11,12,13,14,15);
                }
            }
            // B frag (32x16 f16): lane col, k = {lh*8+0..7, 16+lh*8+0..7}
            {
                const int g0 = ks * 4 + lh;
                const int g1 = g0 + 2;
                #pragma unroll
                for (int t = 0; t < 2; ++t) {
                    const int col = n_base + t * 16 + lr;
                    v8h lo = *(const v8h*)&sB[s][g0][col][0];
                    v8h hi = *(const v8h*)&sB[s][g1][col][0];
                    bf[t] = __builtin_shufflevector(lo, hi,
                        0,1,2,3,4,5,6,7,8,9,10,11,12,13,14,15);
                }
            }
            #pragma unroll
            for (int tm = 0; tm < 2; ++tm) {
                #pragma unroll
                for (int tn = 0; tn < 2; ++tn) {
                    acc[tm][tn] = __builtin_amdgcn_wmma_f32_16x16x32_f16(
                        false, af[tm], false, bf[tn],
                        (short)0, acc[tm][tn], false, false);
                }
            }
        }
    };

    // ---- pipelined main loop: global loads overlap WMMA compute ----
    const int nStages = K / BK;
    load_regs(0);
    store_lds(0);
    __syncthreads();
    for (int st = 0; st < nStages; ++st) {
        const int buf  = st & 1;
        const bool more = (st + 1) < nStages;
        if (more) load_regs((st + 1) * BK);   // issue loads, no wait yet
        compute_stage(buf);                   // 8 WMMAs hide load latency
        if (more) store_lds(buf ^ 1);         // wait + cvt + LDS store
        __syncthreads();
    }

    // ---- epilogue: C/D layout -> global fp32 (lanes 0..15 contiguous in n) ----
    #pragma unroll
    for (int tm = 0; tm < 2; ++tm) {
        #pragma unroll
        for (int tn = 0; tn < 2; ++tn) {
            const int    gm0 = blockM * BM + m_base + tm * 16 + lh * 8;
            const size_t gn  = (size_t)(blockN * BN + n_base + tn * 16 + lr);
            #pragma unroll
            for (int r = 0; r < 8; ++r) {
                C[(size_t)(gm0 + r) * N + gn] = acc[tm][tn][r];
            }
        }
    }
}

extern "C" void kernel_launch(void* const* d_in, const int* in_sizes, int n_in,
                              void* d_out, int out_size, void* d_ws, size_t ws_size,
                              hipStream_t stream) {
    (void)n_in; (void)out_size; (void)d_ws; (void)ws_size;
    const float* A = (const float*)d_in[0];   // inputs [M, K] fp32
    const float* W = (const float*)d_in[1];   // kernel [K, N] fp32
    float*       C = (float*)d_out;           // output [M, N] fp32

    const int K = 1024;
    const int N = 1024;
    const int M = in_sizes[0] / K;            // 16384

    dim3 grid(N / BN, M / BM);                // (8, 128)
    dim3 block(NTHREADS);                     // 16 wave32 waves
    hipLaunchKernelGGL(wage_qgemm_f16_wmma, grid, block, 0, stream,
                       A, W, C, M, N, K);
}
